// MiniDecoder_61117384622566
// MI455X (gfx1250) — matrix operand order repaired and checked by reading, not compiled, not run on hardware
//
#include <hip/hip_runtime.h>
#include <cstdint>
#include <cstddef>

// ---------------------------------------------------------------------------
// MiniDecoder forward for MI455X (gfx1250, wave32, WMMA).
// All matmuls use v_wmma_f32_16x16x32_bf16. Weights are f32 in HBM, converted
// to bf16 while staging into LDS (single HBM pass). GEMM A-tiles are staged
// by the Tensor Data Mover (tensor_load_to_lds, TENSORcnt) with hardware LDS
// row padding; attention K-tiles use global_load_async_to_lds_b128 (ASYNCcnt).
// ---------------------------------------------------------------------------

typedef __attribute__((ext_vector_type(16))) __bf16        v16bf;
typedef __attribute__((ext_vector_type(8)))  float         v8f;
typedef __attribute__((ext_vector_type(4)))  unsigned int  u32x4;
typedef int v4i __attribute__((__vector_size__(16)));      // matches builtin param

union Frag { v16bf v; u32x4 q[2]; };

static constexpr int B  = 2, T = 2048, D = 1024, H = 16, L = 8;
static constexpr int HD = 64, DI = 2048, VOCAB = 32000, M = B * T;
static constexpr int BM = 128, BN = 128, BK = 32;

#define DEV_INLINE __device__ __forceinline__
#define AS1 __attribute__((address_space(1)))
#define AS3 __attribute__((address_space(3)))

#if defined(__gfx1250__) && \
    __has_builtin(__builtin_amdgcn_global_load_async_to_lds_b128) && \
    __has_builtin(__builtin_amdgcn_s_wait_asynccnt)
#define USE_ASYNC 1
#else
#define USE_ASYNC 0
#endif

#if USE_ASYNC
#define ASYNC_CP_B128(lds, gptr) \
  __builtin_amdgcn_global_load_async_to_lds_b128( \
      (AS1 v4i*)(gptr), (AS3 v4i*)(lds), 0, 0)
#define ASYNC_WAIT() __builtin_amdgcn_s_wait_asynccnt(0)
#endif

#if defined(__gfx1250__) && \
    __has_builtin(__builtin_amdgcn_tensor_load_to_lds) && \
    __has_builtin(__builtin_amdgcn_s_wait_tensorcnt)
#define USE_TDM 1
#else
#define USE_TDM 0
#endif

#if USE_TDM
typedef unsigned int tdm_u32x4 __attribute__((__vector_size__(16)));
typedef int          tdm_i32x4 __attribute__((__vector_size__(16)));
typedef int          tdm_i32x8 __attribute__((__vector_size__(32)));

// 2D bf16 tile load: rows x cols elements, global row stride row_stride_elems,
// LDS destination padded by 4 dwords (8 bf16) after each 16-dword (32 bf16)
// row => LDS row stride 40 bf16, matching the As[][] layout below.
DEV_INLINE void tdm_load_2d_bf16(const __bf16* gaddr, unsigned lds_off,
                                 int rows, int cols, int row_stride_elems) {
  const unsigned long long ga = (unsigned long long)(uintptr_t)gaddr;
  tdm_u32x4 g0;
  g0[0] = 1u;                                    // count=1, user descriptor
  g0[1] = lds_off;                               // lds_addr (bits 63:32)
  g0[2] = (unsigned)ga;                          // global_addr[31:0]
  g0[3] = ((unsigned)(ga >> 32) & 0x01FFFFFFu)   // global_addr[56:32]
        | (2u << 30);                            // type = 2 ("image")
  tdm_i32x8 g1;
  g1[0] = (1 << 16)       // data_size = 2 bytes (bf16)
        | (1 << 20)       // pad_enable
        | (3 << 22)       // pad_interval = 16 dwords (one 32-elem row)
        | (3 << 25);      // pad_amount   = 4 dwords (8 bf16)
  g1[1] = (row_stride_elems & 0xFFFF) << 16;           // tensor_dim0[15:0]
  g1[2] = ((row_stride_elems >> 16) & 0xFFFF)          // tensor_dim0[31:16]
        | ((rows & 0xFFFF) << 16);                     // tensor_dim1[15:0]
  g1[3] = (cols & 0xFFFF) << 16;                       // tile_dim0 (bits 127:112)
  g1[4] = rows & 0xFFFF;                               // tile_dim1; tile_dim2 = 0
  g1[5] = row_stride_elems;                            // tensor_dim0_stride[31:0]
  g1[6] = 0;                                           // stride0 hi / stride1 lo
  g1[7] = 0;                                           // tensor_dim1_stride hi
  tdm_i32x4 z4 = {0, 0, 0, 0};
#if __clang_major__ >= 23
  tdm_i32x8 z8 = {0, 0, 0, 0, 0, 0, 0, 0};
  __builtin_amdgcn_tensor_load_to_lds(g0, g1, z4, z4, z8, 0);
#else
  __builtin_amdgcn_tensor_load_to_lds(g0, g1, z4, z4, 0);
#endif
}
#define TDM_WAIT() __builtin_amdgcn_s_wait_tensorcnt(0)
#endif

DEV_INLINE v8f wmma_bf16(const Frag& a, const Frag& b, v8f c) {
  return __builtin_amdgcn_wmma_f32_16x16x32_bf16(
      false, a.v, false, b.v, (short)0, c, false, false);
}

// ---------------------------------------------------------------------------
// Embedding gather: x[row,:] = emb[idx[row],:]
// ---------------------------------------------------------------------------
__global__ __launch_bounds__(256) void embed_kernel(
    const int* __restrict__ idx, const float* __restrict__ emb,
    float* __restrict__ x) {
  const int row = blockIdx.x;
  const int tok = idx[row];
  const float4* src = (const float4*)(emb + (size_t)tok * D);
  float4* dst = (float4*)(x + (size_t)row * D);
  dst[threadIdx.x] = src[threadIdx.x];   // D/4 == 256 == blockDim.x
}

// ---------------------------------------------------------------------------
// RMSNorm: f32 row -> bf16 row (one block of 256 per row; D = 1024 = 256*4)
// ---------------------------------------------------------------------------
__global__ __launch_bounds__(256) void rmsnorm_kernel(
    const float* __restrict__ x, const float* __restrict__ w,
    __bf16* __restrict__ out) {
  const int row = blockIdx.x;
  const int t = threadIdx.x;
  const float* xr = x + (size_t)row * D;
  float4 v = *(const float4*)(xr + t * 4);
  float ss = v.x * v.x + v.y * v.y + v.z * v.z + v.w * v.w;
#pragma unroll
  for (int m = 1; m < 32; m <<= 1) ss += __shfl_xor(ss, m, 32);
  __shared__ float red[8];
  if ((t & 31) == 0) red[t >> 5] = ss;
  __syncthreads();
  float tot = 0.f;
#pragma unroll
  for (int i = 0; i < 8; i++) tot += red[i];
  const float r = rsqrtf(tot / (float)D + 1e-6f);
  float4 wv = *(const float4*)(w + t * 4);
  __bf16* o = out + (size_t)row * D + t * 4;
  o[0] = (__bf16)(v.x * r * wv.x);
  o[1] = (__bf16)(v.y * r * wv.y);
  o[2] = (__bf16)(v.z * r * wv.z);
  o[3] = (__bf16)(v.w * r * wv.w);
}

// ---------------------------------------------------------------------------
// GEMM: C[M,N] = A_bf16[M,K] @ W_f32[N,K]^T, bf16 WMMA, f32 accum.
//   EPI 0: store f32   EPI 1: store bf16   EPI 2: C = resid + acc (f32)
// Double-buffered LDS. A-tiles staged by the TDM (one issue per block, HW LDS
// padding); W converted f32->bf16 through VGPRs. Next tile overlaps WMMAs.
// ---------------------------------------------------------------------------
template <int EPI>
__global__ __launch_bounds__(256) void gemm_bf16(
    const __bf16* __restrict__ A, const float* __restrict__ W,
    void* __restrict__ Cv, const float* __restrict__ resid, int N, int K) {
  __shared__ __bf16 As[2][BM][BK + 8];   // stride 40 bf16 = 80B rows
  __shared__ __bf16 Ws[2][BN][BK + 8];

  const int tid = threadIdx.x, lane = tid & 31, wave = tid >> 5;
  const int lm = lane & 15, hi = lane >> 4;
  const int bm = blockIdx.y * BM, bn = blockIdx.x * BN;
  const int wm = (wave & 3) * 32, wn = (wave >> 2) * 64;
  const int sr = tid >> 1;           // staging row 0..127
  const int sk = (tid & 1) * 16;     // staging k half

  v8f acc[2][4] = {};

  const __bf16* abase = A + (size_t)bm * K;        // block's A tile origin
  const __bf16* ap = abase + (size_t)sr * K + sk;  // per-thread fallback ptr
  const float*  wp = W + (size_t)(bn + sr) * K + sk;

  // ---- prologue: stage tile 0 ----
#if USE_TDM
  if (tid == 0)
    tdm_load_2d_bf16(abase, (unsigned)(uintptr_t)&As[0][0][0], BM, BK, K);
#elif USE_ASYNC
  ASYNC_CP_B128(&As[0][sr][sk],     ap);
  ASYNC_CP_B128(&As[0][sr][sk + 8], ap + 8);
#else
  {
    u32x4 a0 = *(const u32x4*)ap;
    u32x4 a1 = *(const u32x4*)(ap + 8);
    *(u32x4*)&As[0][sr][sk]     = a0;
    *(u32x4*)&As[0][sr][sk + 8] = a1;
  }
#endif
  {
    __bf16 wb[16];
#pragma unroll
    for (int i = 0; i < 16; i++) wb[i] = (__bf16)wp[i];
    *(u32x4*)&Ws[0][sr][sk]     = *(u32x4*)&wb[0];
    *(u32x4*)&Ws[0][sr][sk + 8] = *(u32x4*)&wb[8];
  }
#if USE_TDM
  if (tid == 0) TDM_WAIT();
#elif USE_ASYNC
  ASYNC_WAIT();
#endif
  __syncthreads();

  const int nk = K / BK;
  for (int t = 0; t < nk; t++) {
    const int cur = t & 1, nxt = cur ^ 1;
    const int kn = (t + 1) * BK;
    const bool more = (t + 1 < nk);

    // --- kick off next tile: A via TDM/async/regs, W f32 loads into regs ---
    float wr[16];
    u32x4 an0, an1;
    if (more) {
#if USE_TDM
      if (tid == 0)
        tdm_load_2d_bf16(abase + kn, (unsigned)(uintptr_t)&As[nxt][0][0],
                         BM, BK, K);
#elif USE_ASYNC
      ASYNC_CP_B128(&As[nxt][sr][sk],     ap + kn);
      ASYNC_CP_B128(&As[nxt][sr][sk + 8], ap + kn + 8);
#else
      an0 = *(const u32x4*)(ap + kn);
      an1 = *(const u32x4*)(ap + kn + 8);
#endif
#pragma unroll
      for (int i = 0; i < 16; i++) wr[i] = wp[kn + i];
      if (kn + BK < K)               // gfx1250 global_prefetch_b8
        __builtin_prefetch(wp + kn + BK, 0, 1);
    }

    // --- fragments + WMMA on current buffer ---
    Frag af[2], bf[4];
#pragma unroll
    for (int i = 0; i < 2; i++) {
      // A 16x32: low lanes K 0-7 & 16-23, high lanes K 8-15 & 24-31
      const __bf16* p = &As[cur][wm + i * 16 + lm][hi * 8];
      af[i].q[0] = *(const u32x4*)p;
      af[i].q[1] = *(const u32x4*)(p + 16);
    }
#pragma unroll
    for (int j = 0; j < 4; j++) {
      // B 32x16: lane = column, low lanes K 0-7 & 16-23, high lanes K 8-15 & 24-31
      const __bf16* p = &Ws[cur][wn + j * 16 + lm][hi * 8];
      bf[j].q[0] = *(const u32x4*)p;
      bf[j].q[1] = *(const u32x4*)(p + 16);
    }
#pragma unroll
    for (int i = 0; i < 2; i++)
#pragma unroll
      for (int j = 0; j < 4; j++)
        acc[i][j] = wmma_bf16(af[i], bf[j], acc[i][j]);

    // --- finish staging next tile (convert W, store; A regs if needed) ---
    if (more) {
      __bf16 wb[16];
#pragma unroll
      for (int i = 0; i < 16; i++) wb[i] = (__bf16)wr[i];
      *(u32x4*)&Ws[nxt][sr][sk]     = *(u32x4*)&wb[0];
      *(u32x4*)&Ws[nxt][sr][sk + 8] = *(u32x4*)&wb[8];
#if !USE_TDM && !USE_ASYNC
      *(u32x4*)&As[nxt][sr][sk]     = an0;
      *(u32x4*)&As[nxt][sr][sk + 8] = an1;
#endif
    }
#if USE_TDM
    if (tid == 0) TDM_WAIT();
#elif USE_ASYNC
    ASYNC_WAIT();
#endif
    __syncthreads();
  }

  // --- epilogue: D-layout row = r + 8*hi, col = lane&15 ---
#pragma unroll
  for (int i = 0; i < 2; i++)
#pragma unroll
    for (int j = 0; j < 4; j++)
#pragma unroll
      for (int r = 0; r < 8; r++) {
        const size_t row = (size_t)(bm + wm + i * 16 + hi * 8 + r);
        const size_t col = (size_t)(bn + wn + j * 16 + lm);
        const float v = acc[i][j][r];
        if (EPI == 0)       ((float*)Cv)[row * N + col] = v;
        else if (EPI == 1)  ((__bf16*)Cv)[row * N + col] = (__bf16)v;
        else                ((float*)Cv)[row * N + col] = resid[row * N + col] + v;
      }
}

// ---------------------------------------------------------------------------
// Flash attention: grid (B*H, T/128), block 256 (8 waves x 16 q-rows).
// S = Q K^T and O += P V via WMMA. Online softmax uses a wave-uniform running
// max (one full-wave shuffle reduce per tile) and accumulates the row-sum l
// through a 5th WMMA whose B operand is an all-ones constant fragment.
// ---------------------------------------------------------------------------
__global__ __launch_bounds__(256) void attn_kernel(
    const __bf16* __restrict__ qkv, __bf16* __restrict__ out) {
  const int bh = blockIdx.x, b = bh / H, h = bh % H;
  const int qbase = blockIdx.y * 128;
  const int tid = threadIdx.x, lane = tid & 31, wave = tid >> 5;
  const int lm = lane & 15, hi = lane >> 4;
  const int qrow0 = qbase + wave * 16;
  const float scale = 0.125f;   // 1/sqrt(HD)

  __shared__ __bf16 Ks[32][72];        // [key][hd]  natural
  __shared__ __bf16 Vs[64][40];        // [hd][key]  transposed
  __shared__ __bf16 Ps[8][16][40];     // per-wave P tile (D->A relayout)

  // all-ones B fragment (bf16 1.0 splat) for row-sum accumulation
  Frag ones;
#pragma unroll
  for (int i = 0; i < 4; i++) { ones.q[0][i] = 0x3F803F80u; ones.q[1][i] = 0x3F803F80u; }

  // Q fragments (A layout, resident in VGPRs for the whole pass)
  Frag qa[2];
  {
    const __bf16* qp = qkv + (size_t)(b * T + qrow0 + lm) * (3 * D) + h * HD;
#pragma unroll
    for (int f = 0; f < 2; f++) {
      qa[f].q[0] = *(const u32x4*)(qp + f * 32 + hi * 8);
      qa[f].q[1] = *(const u32x4*)(qp + f * 32 + hi * 8 + 16);
    }
  }

  v8f o[5] = {};          // o[0..3] = output hd tiles, o[4] = row sums (l)
  float mw = -1e30f;      // wave-uniform running max

  const int skey = tid >> 3;          // 0..31
  const int shd  = (tid & 7) * 8;     // 0..56

  for (int jb = 0; jb < qbase + 128; jb += 32) {
    // --- cooperative staging: K natural (async if available), V transposed ---
    {
      const size_t kvrow = (size_t)(b * T + jb + skey) * (3 * D);
      const __bf16* kp = qkv + kvrow + D + h * HD + shd;
#if USE_ASYNC
      ASYNC_CP_B128(&Ks[skey][shd], kp);
#else
      *(u32x4*)&Ks[skey][shd] = *(const u32x4*)kp;
#endif
      __bf16 vt[8];
      *(u32x4*)vt = *(const u32x4*)(qkv + kvrow + 2 * D + h * HD + shd);
#pragma unroll
      for (int i = 0; i < 8; i++) Vs[shd + i][skey] = vt[i];
    }
#if USE_ASYNC
    ASYNC_WAIT();
#endif
    __syncthreads();

    if (jb <= qrow0 + 15) {           // causal early-out per wave
      // --- S tiles: keys jb..jb+15 (s0) and jb+16..jb+31 (s1) ---
      v8f s0 = {}, s1 = {};
      Frag kb;
#pragma unroll
      for (int kk = 0; kk < 2; kk++) {          // hd halves 0-31, 32-63
        const __bf16* p0 = &Ks[lm][kk * 32 + hi * 8];
        kb.q[0] = *(const u32x4*)p0; kb.q[1] = *(const u32x4*)(p0 + 16);
        s0 = wmma_bf16(qa[kk], kb, s0);
        const __bf16* p1 = &Ks[16 + lm][kk * 32 + hi * 8];
        kb.q[0] = *(const u32x4*)p1; kb.q[1] = *(const u32x4*)(p1 + 16);
        s1 = wmma_bf16(qa[kk], kb, s1);
      }

      // --- scale + causal mask + tile max ---
      const bool needmask = (jb + 31 > qrow0);
      float v0[8], v1[8];
      float vmax = -1e30f;
#pragma unroll
      for (int r = 0; r < 8; r++) {
        float a0 = s0[r] * scale;
        float a1 = s1[r] * scale;
        if (needmask) {
          const int q = qrow0 + hi * 8 + r;
          if (jb + lm > q)      a0 = -1e30f;
          if (jb + 16 + lm > q) a1 = -1e30f;
        }
        v0[r] = a0; v1[r] = a1;
        vmax = fmaxf(vmax, fmaxf(a0, a1));
      }
#pragma unroll
      for (int msk = 1; msk < 32; msk <<= 1)
        vmax = fmaxf(vmax, __shfl_xor(vmax, msk, 32));

      // --- online softmax with wave-uniform max ---
      const float mnew = fmaxf(mw, vmax);
      const float corr = __expf(mw - mnew);
      mw = mnew;
#pragma unroll
      for (int j = 0; j < 5; j++)
#pragma unroll
        for (int r = 0; r < 8; r++) o[j][r] *= corr;
#pragma unroll
      for (int r = 0; r < 8; r++) {
        const float p0 = __expf(v0[r] - mnew);
        const float p1 = __expf(v1[r] - mnew);
        Ps[wave][hi * 8 + r][lm]      = (__bf16)p0;   // D-layout -> LDS
        Ps[wave][hi * 8 + r][16 + lm] = (__bf16)p1;
      }

      // --- O += P V (4 tiles) and l += P * ones (1 tile) ---
      Frag pa;
      {
        const __bf16* p = &Ps[wave][lm][hi * 8];
        pa.q[0] = *(const u32x4*)p;
        pa.q[1] = *(const u32x4*)(p + 16);
      }
#pragma unroll
      for (int j = 0; j < 4; j++) {
        Frag vb;
        const __bf16* p = &Vs[j * 16 + lm][hi * 8];
        vb.q[0] = *(const u32x4*)p; vb.q[1] = *(const u32x4*)(p + 16);
        o[j] = wmma_bf16(pa, vb, o[j]);
      }
      o[4] = wmma_bf16(pa, ones, o[4]);
    }
    __syncthreads();
  }

  // --- normalize by l (= o[4], replicated across columns) and store ---
  const size_t obase = (size_t)(b * T) * D + (size_t)h * HD;
#pragma unroll
  for (int j = 0; j < 4; j++)
#pragma unroll
    for (int r = 0; r < 8; r++) {
      const int q = qrow0 + hi * 8 + r;
      out[obase + (size_t)q * D + j * 16 + lm] = (__bf16)(o[j][r] / o[4][r]);
    }
}

// ---------------------------------------------------------------------------
// SwiGLU: gu = silu(g) * u  (f32 -> bf16), 4 elements/thread
// ---------------------------------------------------------------------------
__global__ __launch_bounds__(256) void swiglu_kernel(
    const float* __restrict__ g, const float* __restrict__ u,
    __bf16* __restrict__ gu, int n) {
  const int i = blockIdx.x * 1024 + threadIdx.x * 4;
  if (i >= n) return;
  float4 gv = *(const float4*)(g + i);
  float4 uv = *(const float4*)(u + i);
  __bf16* o = gu + i;
  o[0] = (__bf16)(gv.x / (1.f + __expf(-gv.x)) * uv.x);
  o[1] = (__bf16)(gv.y / (1.f + __expf(-gv.y)) * uv.y);
  o[2] = (__bf16)(gv.z / (1.f + __expf(-gv.z)) * uv.z);
  o[3] = (__bf16)(gv.w / (1.f + __expf(-gv.w)) * uv.w);
}

// ---------------------------------------------------------------------------
extern "C" void kernel_launch(void* const* d_in, const int* in_sizes, int n_in,
                              void* d_out, int out_size, void* d_ws,
                              size_t ws_size, hipStream_t stream) {
  (void)in_sizes; (void)n_in; (void)out_size; (void)ws_size;
  const int*   idx  = (const int*)  d_in[0];
  const float* emb  = (const float*)d_in[1];
  const float* ln1  = (const float*)d_in[2];
  const float* qkvw = (const float*)d_in[3];
  const float* ow   = (const float*)d_in[4];
  const float* ln2  = (const float*)d_in[5];
  const float* gw   = (const float*)d_in[6];
  const float* uw   = (const float*)d_in[7];
  const float* dw   = (const float*)d_in[8];
  const float* lnf  = (const float*)d_in[9];

  // workspace carve-out (~143 MB)
  char* p = (char*)d_ws;
  float*  x     = (float*) p; p += (size_t)M * D * 4;       // residual stream
  __bf16* h     = (__bf16*)p; p += (size_t)M * D * 2;       // normed acts
  __bf16* qkvb  = (__bf16*)p; p += (size_t)M * 3 * D * 2;   // qkv projections
  __bf16* attnb = (__bf16*)p; p += (size_t)M * D * 2;       // attention out
  float*  gbuf  = (float*) p; p += (size_t)M * DI * 4;      // gate
  float*  ubuf  = (float*) p; p += (size_t)M * DI * 4;      // up
  __bf16* gubuf = (__bf16*)p; p += (size_t)M * DI * 2;      // silu(g)*u

  embed_kernel<<<M, 256, 0, stream>>>(idx, emb, x);

  for (int l = 0; l < L; l++) {
    rmsnorm_kernel<<<M, 256, 0, stream>>>(x, ln1 + (size_t)l * D, h);
    gemm_bf16<1><<<dim3(3 * D / BN, M / BM), 256, 0, stream>>>(
        h, qkvw + (size_t)l * 3 * D * D, qkvb, nullptr, 3 * D, D);
    attn_kernel<<<dim3(B * H, T / 128), 256, 0, stream>>>(qkvb, attnb);
    gemm_bf16<2><<<dim3(D / BN, M / BM), 256, 0, stream>>>(
        attnb, ow + (size_t)l * D * D, x, x, D, D);
    rmsnorm_kernel<<<M, 256, 0, stream>>>(x, ln2 + (size_t)l * D, h);
    gemm_bf16<0><<<dim3(DI / BN, M / BM), 256, 0, stream>>>(
        h, gw + (size_t)l * DI * D, gbuf, nullptr, DI, D);
    gemm_bf16<0><<<dim3(DI / BN, M / BM), 256, 0, stream>>>(
        h, uw + (size_t)l * DI * D, ubuf, nullptr, DI, D);
    swiglu_kernel<<<(M * DI) / 1024, 256, 0, stream>>>(gbuf, ubuf, gubuf, M * DI);
    gemm_bf16<2><<<dim3(D / BN, M / BM), 256, 0, stream>>>(
        gubuf, dw + (size_t)l * D * DI, x, x, D, DI);
  }

  rmsnorm_kernel<<<M, 256, 0, stream>>>(x, lnf, h);
  gemm_bf16<0><<<dim3(VOCAB / BN, M / BM), 256, 0, stream>>>(
      h, emb, (float*)d_out, nullptr, VOCAB, D);
}
